// MambaBlock_50027779063973
// MI455X (gfx1250) — matrix operand order repaired and checked by reading, not compiled, not run on hardware
//
#include <hip/hip_runtime.h>
#include <hip/hip_bf16.h>
#include <math.h>

typedef __attribute__((ext_vector_type(16))) _Float16 v16h;
typedef __attribute__((ext_vector_type(8)))  _Float16 v8h;
typedef __attribute__((ext_vector_type(8)))  float    v8f;

#define B_SZ    2
#define T_SZ    4096
#define DMODEL  1024
#define DSTATE  128
#define DCONV   4
#define DINNER  2048
#define HEADDIM 64
#define NHEADS  32
#define DINPROJ 4384      // 2*DINNER + 2*DSTATE + NHEADS
#define CONVDIM 2304      // DINNER + 2*DSTATE
#define CHUNK   64        // SSD chunk length

// ---------------------------------------------------------------------------
// WMMA fragment helpers (LDS rows are [row][k], 16-byte aligned, stride%8==0)
// A operand (m-side): lane L holds m=L%16, k in {g*8..g*8+7, 16+g*8..+7}, g=L/16
// B operand (n-side): lane L holds n=L%16, k in {g*8..g*8+7, 16+g*8..+7}, g=L/16
// ---------------------------------------------------------------------------
__device__ __forceinline__ v16h frag_m(const _Float16* rowp, int g) {
  v8h lo = *(const v8h*)(rowp + g * 8);
  v8h hi = *(const v8h*)(rowp + 16 + g * 8);
  return __builtin_shufflevector(lo, hi, 0, 1, 2, 3, 4, 5, 6, 7,
                                         8, 9, 10, 11, 12, 13, 14, 15);
}
__device__ __forceinline__ v16h frag_n(const _Float16* rowp, int g) {
  v8h lo = *(const v8h*)(rowp + g * 8);
  v8h hi = *(const v8h*)(rowp + 16 + g * 8);
  return __builtin_shufflevector(lo, hi, 0, 1, 2, 3, 4, 5, 6, 7,
                                         8, 9, 10, 11, 12, 13, 14, 15);
}
__device__ __forceinline__ v8f wmma_step(v8f acc, v16h a, v16h b) {
  return __builtin_amdgcn_wmma_f32_16x16x32_f16(false, a, false, b,
                                                (short)0, acc, false, false);
}

// ---------------------------------------------------------------------------
// elementwise f32 -> f16 convert (for weights)
// ---------------------------------------------------------------------------
__global__ void cvt_f16_kernel(const float* __restrict__ src, _Float16* __restrict__ dst, int n) {
  int i = blockIdx.x * 256 + threadIdx.x;
  if (i < n) dst[i] = (_Float16)src[i];
}

// ---------------------------------------------------------------------------
// LayerNorm over D_MODEL; input x is (B, D_MODEL, T); output xn (B*T, D_MODEL) f16
// ---------------------------------------------------------------------------
__global__ __launch_bounds__(256) void ln_kernel(const float* __restrict__ x,
                                                 const float* __restrict__ ln_w,
                                                 const float* __restrict__ ln_b,
                                                 _Float16* __restrict__ xn) {
  int row = blockIdx.x;                 // b*T + t
  int b = row / T_SZ, t = row % T_SZ;
  const float* xp = x + (size_t)b * DMODEL * T_SZ + t;   // stride T between d
  __shared__ float red[256];
  __shared__ float red2[256];
  float v[4];
  float s = 0.f, s2 = 0.f;
  #pragma unroll
  for (int i = 0; i < 4; ++i) {
    int d = threadIdx.x + i * 256;
    float val = xp[(size_t)d * T_SZ];
    v[i] = val; s += val; s2 += val * val;
  }
  red[threadIdx.x] = s; red2[threadIdx.x] = s2;
  __syncthreads();
  for (int off = 128; off > 0; off >>= 1) {
    if (threadIdx.x < off) { red[threadIdx.x] += red[threadIdx.x + off]; red2[threadIdx.x] += red2[threadIdx.x + off]; }
    __syncthreads();
  }
  float mu = red[0] * (1.f / DMODEL);
  float var = red2[0] * (1.f / DMODEL) - mu * mu;
  float rinv = rsqrtf(var + 1e-5f);
  #pragma unroll
  for (int i = 0; i < 4; ++i) {
    int d = threadIdx.x + i * 256;
    float o = (v[i] - mu) * rinv * ln_w[d] + ln_b[d];
    xn[(size_t)row * DMODEL + d] = (_Float16)o;
  }
}

// ---------------------------------------------------------------------------
// WMMA GEMM: C[m,n] = sum_k A[m,k] * Bw[n,k]   (A: MxK f16, Bw: NxK f16)
// 256 threads = 8 waves; wave grid 4(M) x 2(N); per wave 2x4 tiles of 16x16.
// Block tile 128x128, K-step 32, double-buffered LDS.
// out_mode 0: C row-major MxN; out_mode 1: C[((m/T)*N + n)*T + m%T] (B,D,T)
// ---------------------------------------------------------------------------
__global__ __launch_bounds__(256) void wmma_gemm_kernel(const _Float16* __restrict__ A,
                                                        const _Float16* __restrict__ Bw,
                                                        float* __restrict__ C,
                                                        int M, int N, int K, int out_mode) {
  __shared__ __align__(16) _Float16 As[2][128][40];   // [buf][m][k] (+pad, 80B row)
  __shared__ __align__(16) _Float16 Bs[2][128][40];   // [buf][n][k]

  int tid  = threadIdx.x;
  int lane = tid & 31;
  int w    = tid >> 5;        // 0..7
  int wm   = w & 3;           // 4 waves along M
  int wn   = w >> 2;          // 2 waves along N
  int m0   = blockIdx.x * 128;
  int n0   = blockIdx.y * 128;

  int sr   = tid >> 1;            // staging row 0..127
  int koff = (tid & 1) * 16;      // staging k offset (two uint4 chunks)

  const _Float16* aRow = A + (size_t)(m0 + sr) * K + koff;
  int gn = n0 + sr; if (gn > N - 1) gn = N - 1;          // clamp: pad cols harmless
  const _Float16* bRow = Bw + (size_t)gn * K + koff;

  v8f acc[2][4];
  v8f zero = {0.f, 0.f, 0.f, 0.f, 0.f, 0.f, 0.f, 0.f};
  #pragma unroll
  for (int i = 0; i < 2; ++i)
    #pragma unroll
    for (int j = 0; j < 4; ++j) acc[i][j] = zero;

  int mloc = lane & 15;
  int g    = lane >> 4;       // 0 or 1

  const int NK = K >> 5;      // K / 32

  { // prologue: stage tile 0 into buffer 0
    uint4 a0 = *(const uint4*)(aRow);
    uint4 a1 = *(const uint4*)(aRow + 8);
    uint4 b0 = *(const uint4*)(bRow);
    uint4 b1 = *(const uint4*)(bRow + 8);
    *(uint4*)&As[0][sr][koff]     = a0;
    *(uint4*)&As[0][sr][koff + 8] = a1;
    *(uint4*)&Bs[0][sr][koff]     = b0;
    *(uint4*)&Bs[0][sr][koff + 8] = b1;
  }
  __syncthreads();

  for (int kt = 0; kt < NK; ++kt) {
    int cur = kt & 1;
    uint4 a0, a1, b0, b1;
    bool more = (kt + 1 < NK);
    if (more) {
      const _Float16* ap = aRow + (size_t)(kt + 1) * 32;
      const _Float16* bp = bRow + (size_t)(kt + 1) * 32;
      a0 = *(const uint4*)(ap);
      a1 = *(const uint4*)(ap + 8);
      b0 = *(const uint4*)(bp);
      b1 = *(const uint4*)(bp + 8);
    }
    if (kt + 2 < NK) {
      __builtin_prefetch(aRow + (size_t)(kt + 2) * 32, 0, 1);
      __builtin_prefetch(bRow + (size_t)(kt + 2) * 32, 0, 1);
    }

    v16h afr[2], bfr[4];
    #pragma unroll
    for (int i = 0; i < 2; ++i)
      afr[i] = frag_m(&As[cur][wm * 32 + i * 16 + mloc][0], g);
    #pragma unroll
    for (int j = 0; j < 4; ++j)
      bfr[j] = frag_n(&Bs[cur][wn * 64 + j * 16 + mloc][0], g);

    #pragma unroll
    for (int i = 0; i < 2; ++i)
      #pragma unroll
      for (int j = 0; j < 4; ++j)
        acc[i][j] = wmma_step(acc[i][j], afr[i], bfr[j]);

    if (more) {
      int nxt = cur ^ 1;
      *(uint4*)&As[nxt][sr][koff]     = a0;
      *(uint4*)&As[nxt][sr][koff + 8] = a1;
      *(uint4*)&Bs[nxt][sr][koff]     = b0;
      *(uint4*)&Bs[nxt][sr][koff + 8] = b1;
      __syncthreads();
    }
  }

  #pragma unroll
  for (int i = 0; i < 2; ++i)
    #pragma unroll
    for (int j = 0; j < 4; ++j)
      #pragma unroll
      for (int vv = 0; vv < 8; ++vv) {
        int m = m0 + wm * 32 + i * 16 + vv + 8 * g;
        int n = n0 + wn * 64 + j * 16 + mloc;
        if (n < N) {
          float val = acc[i][j][vv];
          if (out_mode == 0) {
            C[(size_t)m * N + n] = val;
          } else {
            int b = m / T_SZ, t = m % T_SZ;
            C[((size_t)b * N + n) * T_SZ + t] = val;
          }
        }
      }
}

// ---------------------------------------------------------------------------
// dt = softplus(raw + dt_bias)
// ---------------------------------------------------------------------------
__global__ void dt_kernel(const float* __restrict__ zxbcdt,
                          const float* __restrict__ dt_bias,
                          float* __restrict__ dtArr) {
  int i = blockIdx.x * 256 + threadIdx.x;     // over B*T*NHEADS
  if (i >= B_SZ * T_SZ * NHEADS) return;
  int h = i & (NHEADS - 1);
  int bt = i >> 5;
  float v = zxbcdt[(size_t)bt * DINPROJ + (DINNER + CONVDIM) + h] + dt_bias[h];
  dtArr[i] = (v > 20.f) ? v : log1pf(expf(v));
}

// ---------------------------------------------------------------------------
// depthwise causal conv(4) + bias + SiLU over xBC slice of zxbcdt
// ---------------------------------------------------------------------------
__global__ void conv_kernel(const float* __restrict__ zxbcdt,
                            const float* __restrict__ conv_w,
                            const float* __restrict__ conv_b,
                            float* __restrict__ xbc) {
  long i = (long)blockIdx.x * 256 + threadIdx.x;       // over B*T*CONVDIM
  if (i >= (long)B_SZ * T_SZ * CONVDIM) return;
  int c = (int)(i % CONVDIM);
  long bt = i / CONVDIM;
  int t = (int)(bt % T_SZ);
  long b = bt / T_SZ;
  float acc = conv_b[c];
  #pragma unroll
  for (int k = 0; k < DCONV; ++k) {
    int tt = t - (DCONV - 1) + k;
    if (tt >= 0)
      acc += conv_w[c * DCONV + k] * zxbcdt[((size_t)(b * T_SZ + tt)) * DINPROJ + DINNER + c];
  }
  float s = acc * (1.f / (1.f + expf(-acc)));
  xbc[(size_t)bt * CONVDIM + c] = s;
}

// ---------------------------------------------------------------------------
// Chunked SSD scan (Mamba-2): one block per (b,h), chunk L=64.
// Per chunk (all WMMA, f32 accumulate, operands f16 in LDS):
//   G = C*B^T (LxL,K=128);  M[t,s] = G * exp(S[t]-S[s]) * [t>=s]
//   Y = M*Xdt (K=64) + diag(exp(S)) * (C*h_in^T) (K=128) + D*xin
//   h_out = exp(S_last)*h_in + (Xdt^T * diag(w)) * B  (64x128, K=64)
// where S = inclusive prefix sum of dt*A (A<0), w[s]=exp(S_last - S[s]).
// State h (64x128) persists in LDS across the 64 chunks.
// ---------------------------------------------------------------------------
__global__ __launch_bounds__(256) void ssd_chunk_kernel(const float* __restrict__ xbc,
                                                        const float* __restrict__ dtArr,
                                                        const float* __restrict__ A_log,
                                                        const float* __restrict__ Dp,
                                                        const float* __restrict__ h0,
                                                        float* __restrict__ y,
                                                        float* __restrict__ hOut) {
  __shared__ __align__(16) _Float16 Bsm[CHUNK][136];    // B [s][n]
  __shared__ __align__(16) _Float16 Csm[CHUNK][136];    // C [t][n]
  __shared__ __align__(16) _Float16 BsT[DSTATE][72];    // B^T [n][s]
  __shared__ __align__(16) _Float16 h16[HEADDIM][136];  // h (f16) [p][n]
  __shared__ __align__(16) _Float16 XdtT[HEADDIM][72];  // xin*dt [p][s]
  __shared__ __align__(16) _Float16 XdtW[HEADDIM][72];  // xin*dt*w [p][s]
  __shared__ __align__(16) _Float16 Msm[CHUNK][72];     // decay-masked G [t][s]
  __shared__ __align__(16) float hF[HEADDIM][DSTATE];   // h (f32) [p][n]
  __shared__ __align__(16) float Xin[CHUNK][HEADDIM];   // xin [t][p]
  __shared__ float dts_sh[CHUNK], Ssh[CHUNK], expS_sh[CHUNK], wsh[CHUNK];

  int bh = blockIdx.x;              // b*NHEADS + h
  int b = bh >> 5, h = bh & 31;
  int tid = threadIdx.x, lane = tid & 31, w = tid >> 5;
  int mloc = lane & 15, g = lane >> 4;
  float Ah = -expf(A_log[h]);
  float Dv = Dp[h];
  v8f zero = {0.f, 0.f, 0.f, 0.f, 0.f, 0.f, 0.f, 0.f};

  // init state from h0
  #pragma unroll
  for (int j = 0; j < 8; ++j) {                 // 2048 float4
    int idx4 = tid + j * 256;
    int flat = idx4 * 4;
    int p = flat >> 7, n = flat & 127;
    float4 v = *(const float4*)(h0 + (size_t)bh * HEADDIM * DSTATE + flat);
    *(float4*)&hF[p][n] = v;
    h16[p][n]     = (_Float16)v.x;
    h16[p][n + 1] = (_Float16)v.y;
    h16[p][n + 2] = (_Float16)v.z;
    h16[p][n + 3] = (_Float16)v.w;
  }
  __syncthreads();

  for (int c = 0; c < T_SZ / CHUNK; ++c) {
    size_t btBase = (size_t)b * T_SZ + (size_t)c * CHUNK;

    // ---- stage chunk inputs ----
    if (tid < CHUNK) dts_sh[tid] = dtArr[(btBase + tid) * NHEADS + h];
    #pragma unroll
    for (int j = 0; j < 16; ++j) {              // B,C: 64 rows x 64 float4
      int idx = tid + j * 256;
      int s = idx >> 6, c4 = idx & 63;
      float4 v = *(const float4*)(xbc + (btBase + s) * CONVDIM + DINNER + c4 * 4);
      int col = c4 * 4;
      if (c4 < 32) {
        Bsm[s][col]     = (_Float16)v.x;  BsT[col][s]     = (_Float16)v.x;
        Bsm[s][col + 1] = (_Float16)v.y;  BsT[col + 1][s] = (_Float16)v.y;
        Bsm[s][col + 2] = (_Float16)v.z;  BsT[col + 2][s] = (_Float16)v.z;
        Bsm[s][col + 3] = (_Float16)v.w;  BsT[col + 3][s] = (_Float16)v.w;
      } else {
        Csm[s][col - 128]     = (_Float16)v.x;
        Csm[s][col - 128 + 1] = (_Float16)v.y;
        Csm[s][col - 128 + 2] = (_Float16)v.z;
        Csm[s][col - 128 + 3] = (_Float16)v.w;
      }
    }
    #pragma unroll
    for (int j = 0; j < 4; ++j) {               // xin: 64 rows x 16 float4
      int idx = tid + j * 256;
      int s = idx >> 4, p4 = idx & 15;
      float4 v = *(const float4*)(xbc + (btBase + s) * CONVDIM + h * HEADDIM + p4 * 4);
      *(float4*)&Xin[s][p4 * 4] = v;
    }
    __syncthreads();

    // ---- prefix sum of log-decay: S[t] = sum_{u<=t} dt[u]*A ----
    if (tid < CHUNK) Ssh[tid] = dts_sh[tid] * Ah;
    __syncthreads();
    for (int d = 1; d < CHUNK; d <<= 1) {
      float tv = 0.f;
      if (tid < CHUNK && tid >= d) tv = Ssh[tid - d];
      __syncthreads();
      if (tid < CHUNK) Ssh[tid] += tv;
      __syncthreads();
    }
    if (tid < CHUNK) {
      expS_sh[tid] = expf(Ssh[tid]);
      wsh[tid] = expf(Ssh[CHUNK - 1] - Ssh[tid]);
    }
    __syncthreads();

    // ---- build Xdt operands ----
    #pragma unroll
    for (int j = 0; j < 16; ++j) {              // 4096 elements [p][s]
      int idx = tid + j * 256;
      int p = idx >> 6, s = idx & 63;
      float xv = Xin[s][p] * dts_sh[s];
      XdtT[p][s] = (_Float16)xv;
      XdtW[p][s] = (_Float16)(xv * wsh[s]);
    }
    __syncthreads();

    // ---- GEMM1: G = C*B^T, then decay-mask into Ms ----
    #pragma unroll
    for (int i = 0; i < 2; ++i) {
      int tile = w * 2 + i;
      int gt = tile >> 2, gs = tile & 3;
      v8f acc = zero;
      #pragma unroll
      for (int ks = 0; ks < 4; ++ks)
        acc = wmma_step(acc, frag_m(&Csm[gt * 16 + mloc][ks * 32], g),
                             frag_n(&Bsm[gs * 16 + mloc][ks * 32], g));
      #pragma unroll
      for (int vv = 0; vv < 8; ++vv) {
        int t = gt * 16 + vv + 8 * g;
        int s = gs * 16 + mloc;
        float mv = (t >= s) ? acc[vv] * expf(Ssh[t] - Ssh[s]) : 0.f;
        Msm[t][s] = (_Float16)mv;
      }
    }
    __syncthreads();

    // ---- Y tiles: Y = diag(expS)*(C*h^T) + M*Xdt + D*xin ----
    #pragma unroll
    for (int i = 0; i < 2; ++i) {
      int tile = w * 2 + i;
      int yt = tile >> 2, yp = tile & 3;
      v8f acc = zero;
      #pragma unroll
      for (int ks = 0; ks < 4; ++ks)
        acc = wmma_step(acc, frag_m(&Csm[yt * 16 + mloc][ks * 32], g),
                             frag_n(&h16[yp * 16 + mloc][ks * 32], g));
      #pragma unroll
      for (int vv = 0; vv < 8; ++vv)
        acc[vv] *= expS_sh[yt * 16 + vv + 8 * g];
      #pragma unroll
      for (int ks = 0; ks < 2; ++ks)
        acc = wmma_step(acc, frag_m(&Msm[yt * 16 + mloc][ks * 32], g),
                             frag_n(&XdtT[yp * 16 + mloc][ks * 32], g));
      #pragma unroll
      for (int vv = 0; vv < 8; ++vv) {
        int t = yt * 16 + vv + 8 * g;
        int p = yp * 16 + mloc;
        float yv = acc[vv] + Dv * Xin[t][p];
        y[(btBase + t) * DINNER + h * HEADDIM + p] = yv;
      }
    }

    // ---- state update: h = exp(S_last)*h + (XdtW)*(B) ----
    v8f hacc[4];
    #pragma unroll
    for (int i = 0; i < 4; ++i) {
      int tile = w * 4 + i;
      int pt = tile >> 3, nt = tile & 7;
      v8f acc = zero;
      #pragma unroll
      for (int ks = 0; ks < 2; ++ks)
        acc = wmma_step(acc, frag_m(&XdtW[pt * 16 + mloc][ks * 32], g),
                             frag_n(&BsT[nt * 16 + mloc][ks * 32], g));
      hacc[i] = acc;
    }
    __syncthreads();
    float dtot = expf(Ssh[CHUNK - 1]);
    #pragma unroll
    for (int i = 0; i < 4; ++i) {
      int tile = w * 4 + i;
      int pt = tile >> 3, nt = tile & 7;
      #pragma unroll
      for (int vv = 0; vv < 8; ++vv) {
        int p = pt * 16 + vv + 8 * g;
        int n = nt * 16 + mloc;
        float hnew = hacc[i][vv] + dtot * hF[p][n];
        hF[p][n] = hnew;
        h16[p][n] = (_Float16)hnew;
      }
    }
    __syncthreads();
  }

  // write final hidden state
  #pragma unroll
  for (int j = 0; j < 8; ++j) {
    int idx4 = tid + j * 256;
    int flat = idx4 * 4;
    int p = flat >> 7, n = flat & 127;
    *(float4*)(hOut + (size_t)bh * HEADDIM * DSTATE + flat) = *(float4*)&hF[p][n];
  }
}

// ---------------------------------------------------------------------------
// g = y * silu(z); RMSNorm over D_INNER; write g as f16 for out-proj GEMM
// ---------------------------------------------------------------------------
__global__ __launch_bounds__(256) void gate_kernel(const float* __restrict__ y,
                                                   const float* __restrict__ zxbcdt,
                                                   const float* __restrict__ rms_w,
                                                   _Float16* __restrict__ gOut) {
  int row = blockIdx.x;            // b*T + t
  const float* yp = y + (size_t)row * DINNER;
  const float* zp = zxbcdt + (size_t)row * DINPROJ;   // z = first DINNER cols
  float gv[8];
  float s2 = 0.f;
  #pragma unroll
  for (int i = 0; i < 8; ++i) {
    int c = threadIdx.x + i * 256;
    float z = zp[c];
    float val = yp[c] * (z / (1.f + expf(-z)));
    gv[i] = val; s2 += val * val;
  }
  __shared__ float red[256];
  red[threadIdx.x] = s2;
  __syncthreads();
  for (int off = 128; off > 0; off >>= 1) {
    if (threadIdx.x < off) red[threadIdx.x] += red[threadIdx.x + off];
    __syncthreads();
  }
  float rinv = rsqrtf(red[0] * (1.f / DINNER) + 1e-5f);
  #pragma unroll
  for (int i = 0; i < 8; ++i) {
    int c = threadIdx.x + i * 256;
    gOut[(size_t)row * DINNER + c] = (_Float16)(gv[i] * rinv * rms_w[c]);
  }
}

// ---------------------------------------------------------------------------
extern "C" void kernel_launch(void* const* d_in, const int* in_sizes, int n_in,
                              void* d_out, int out_size, void* d_ws, size_t ws_size,
                              hipStream_t stream) {
  const float* x       = (const float*)d_in[0];
  const float* ln_w    = (const float*)d_in[1];
  const float* ln_b    = (const float*)d_in[2];
  const float* W_in    = (const float*)d_in[3];
  const float* conv_w  = (const float*)d_in[4];
  const float* conv_b  = (const float*)d_in[5];
  const float* dt_bias = (const float*)d_in[6];
  const float* A_log   = (const float*)d_in[7];
  const float* Dp      = (const float*)d_in[8];
  const float* rms_w   = (const float*)d_in[9];
  const float* W_out   = (const float*)d_in[10];
  const float* h0      = (const float*)d_in[11];

  char* ws = (char*)d_ws;
  size_t off = 0;
  auto alloc = [&](size_t bytes) { size_t o = off; off += (bytes + 255) & ~(size_t)255; return o; };

  const size_t MROWS = (size_t)B_SZ * T_SZ;                       // 8192
  _Float16* xn_f16   = (_Float16*)(ws + alloc(MROWS * DMODEL * 2));
  _Float16* Win_f16  = (_Float16*)(ws + alloc((size_t)DINPROJ * DMODEL * 2));
  _Float16* Wout_f16 = (_Float16*)(ws + alloc((size_t)DMODEL * DINNER * 2));
  float*    zxbcdt   = (float*)   (ws + alloc(MROWS * DINPROJ * 4));
  float*    xbc      = (float*)   (ws + alloc(MROWS * CONVDIM * 4));
  float*    dtArr    = (float*)   (ws + alloc(MROWS * NHEADS * 4));
  float*    yBuf     = (float*)   (ws + alloc(MROWS * DINNER * 4));
  _Float16* g_f16    = (_Float16*)(ws + alloc(MROWS * DINNER * 2));

  float* out_main = (float*)d_out;                                // (B, D_MODEL, T)
  float* hOut     = out_main + (size_t)B_SZ * DMODEL * T_SZ;      // (B,NHEADS,HEADDIM,DSTATE)

  {
    int n1 = DINPROJ * DMODEL;
    cvt_f16_kernel<<<(n1 + 255) / 256, 256, 0, stream>>>(W_in, Win_f16, n1);
    int n2 = DMODEL * DINNER;
    cvt_f16_kernel<<<(n2 + 255) / 256, 256, 0, stream>>>(W_out, Wout_f16, n2);
  }
  ln_kernel<<<(int)MROWS, 256, 0, stream>>>(x, ln_w, ln_b, xn_f16);
  {
    dim3 grid((int)(MROWS / 128), (DINPROJ + 127) / 128);
    wmma_gemm_kernel<<<grid, 256, 0, stream>>>(xn_f16, Win_f16, zxbcdt,
                                               (int)MROWS, DINPROJ, DMODEL, 0);
  }
  {
    int n = B_SZ * T_SZ * NHEADS;
    dt_kernel<<<(n + 255) / 256, 256, 0, stream>>>(zxbcdt, dt_bias, dtArr);
  }
  {
    long n = (long)B_SZ * T_SZ * CONVDIM;
    conv_kernel<<<(int)((n + 255) / 256), 256, 0, stream>>>(zxbcdt, conv_w, conv_b, xbc);
  }
  ssd_chunk_kernel<<<B_SZ * NHEADS, 256, 0, stream>>>(xbc, dtArr, A_log, Dp, h0, yBuf, hOut);
  gate_kernel<<<(int)MROWS, 256, 0, stream>>>(yBuf, zxbcdt, rms_w, g_f16);
  {
    dim3 grid((int)(MROWS / 128), DMODEL / 128);
    wmma_gemm_kernel<<<grid, 256, 0, stream>>>(g_f16, Wout_f16, out_main,
                                               (int)MROWS, DMODEL, DINNER, 1);
  }
}
